// VanillaRNN_65292092834055
// MI455X (gfx1250) — compile-verified, matrix operand order checked by reading, not run on hardware
//
#include <hip/hip_runtime.h>

// ---------------- problem constants ----------------
constexpr int B_ = 512;   // batch
constexpr int S_ = 256;   // sequence length
constexpr int V_ = 128;   // vocab
constexpr int E_ = 128;   // embed dim
constexpr int H_ = 512;   // hidden dim

typedef __attribute__((ext_vector_type(16))) __bf16 v16bf;
typedef __attribute__((ext_vector_type(8)))  __bf16 v8bf;
typedef __attribute__((ext_vector_type(8)))  float  v8f;

// LDS staging layout (bf16 elements) for the scan kernel's per-WG weight slices:
//   sWih0: KT=4,  4 frags/kt ->  8192 elems (16 KB)
//   sWhh0: KT=16, 4 frags/kt -> 32768 elems (64 KB)
//   sWih1: KT=16              -> 32768 elems (64 KB)
//   sWhh1: KT=16              -> 32768 elems (64 KB)
constexpr int LDS_WIH0_OFF = 0;
constexpr int LDS_WHH0_OFF = 8192;
constexpr int LDS_WIH1_OFF = 8192 + 32768;
constexpr int LDS_WHH1_OFF = 8192 + 2 * 32768;
constexpr int LDS_ELEMS    = 8192 + 3 * 32768;            // 106496 elems
constexpr size_t LDS_BYTES = (size_t)LDS_ELEMS * 2;       // 208 KB (< 320 KB WGP LDS)

// ---------------- helpers ----------------
__device__ __forceinline__ __bf16 to_bf16(float f) {
    union { float f; unsigned u; } v; v.f = f;
    unsigned u = v.u + 0x7FFFu + ((v.u >> 16) & 1u);   // round-to-nearest-even
    unsigned short h = (unsigned short)(u >> 16);
    __bf16 r;
    __builtin_memcpy(&r, &h, sizeof(h));
    return r;
}

__device__ __forceinline__ v8f wmma_bf16(v16bf a, v16bf b, v8f c) {
    // (neg_a, A, neg_b, B, c_mod, C, reuse_a, reuse_b)
    return __builtin_amdgcn_wmma_f32_16x16x32_bf16(false, a, false, b, (short)0, c,
                                                   false, false);
}

// Load one 16x32 bf16 A-fragment from a row-major matrix.
// ISA layout (16-bit A 16x32): lane L handles row M = L%16;
//   lanes 0-15:  VGPR0-3 = K 0..7,  VGPR4-7 = K 16..23
//   lanes 16-31: VGPR0-3 = K 8..15, VGPR4-7 = K 24..31
__device__ __forceinline__ v16bf load_a_row(const __bf16* __restrict__ rowp, int k0, int lane) {
    const __bf16* p = rowp + k0 + ((lane >> 4) << 3);
    v8bf lo = *(const v8bf*)p;
    v8bf hi = *(const v8bf*)(p + 16);
    v16bf a;
#pragma unroll
    for (int i = 0; i < 8; ++i) { a[i] = lo[i]; a[i + 8] = hi[i]; }
    return a;
}

// B fragment from global pre-packed weights (lane-major, 1 KB / fragment).
__device__ __forceinline__ v16bf load_b_frag(const __bf16* __restrict__ packed,
                                             int NT, int kt, int nt, int lane) {
    const __bf16* p = packed + ((((size_t)kt * NT + nt) * 32 + lane) << 4);
    v8bf lo = *(const v8bf*)p;
    v8bf hi = *(const v8bf*)(p + 8);
    v16bf b;
#pragma unroll
    for (int i = 0; i < 8; ++i) { b[i] = lo[i]; b[i + 8] = hi[i]; }
    return b;
}

// B fragment from an LDS-staged slice (4 fragments per kt): ds_load path.
__device__ __forceinline__ v16bf load_b_frag_lds(const __bf16* lds, int kt, int j, int lane) {
    const __bf16* p = lds + (((((kt << 2) + j) << 5) + lane) << 4);
    v8bf lo = *(const v8bf*)p;
    v8bf hi = *(const v8bf*)(p + 8);
    v16bf b;
#pragma unroll
    for (int i = 0; i < 8; ++i) { b[i] = lo[i]; b[i + 8] = hi[i]; }
    return b;
}

// Fully-unrolled K-loop with all 4 B fragments live per kt:
// forces distinct VGPRs -> many outstanding ds_loads, lets the scheduler
// software-pipeline iteration kt+1 loads above iteration kt WMMAs.
template <int KT>
__device__ __forceinline__ void gemm_acc_lds(const __bf16* __restrict__ arow,
                                             const __bf16* sW, int lane, v8f c[4]) {
#pragma unroll
    for (int kt = 0; kt < KT; ++kt) {
        v16bf a  = load_a_row(arow, kt * 32, lane);
        v16bf b0 = load_b_frag_lds(sW, kt, 0, lane);
        v16bf b1 = load_b_frag_lds(sW, kt, 1, lane);
        v16bf b2 = load_b_frag_lds(sW, kt, 2, lane);
        v16bf b3 = load_b_frag_lds(sW, kt, 3, lane);
        c[0] = wmma_bf16(a, b0, c[0]);
        c[1] = wmma_bf16(a, b1, c[1]);
        c[2] = wmma_bf16(a, b2, c[2]);
        c[3] = wmma_bf16(a, b3, c[3]);
    }
}

// Same structure with global-memory packed B (output projection).
template <int KT>
__device__ __forceinline__ void gemm_acc_glb(const __bf16* __restrict__ arow,
                                             const __bf16* __restrict__ packed,
                                             int NT, int nt0, int lane, v8f c[4]) {
#pragma unroll
    for (int kt = 0; kt < KT; ++kt) {
        v16bf a  = load_a_row(arow, kt * 32, lane);
        v16bf b0 = load_b_frag(packed, NT, kt, nt0 + 0, lane);
        v16bf b1 = load_b_frag(packed, NT, kt, nt0 + 1, lane);
        v16bf b2 = load_b_frag(packed, NT, kt, nt0 + 2, lane);
        v16bf b3 = load_b_frag(packed, NT, kt, nt0 + 3, lane);
        c[0] = wmma_bf16(a, b0, c[0]);
        c[1] = wmma_bf16(a, b1, c[1]);
        c[2] = wmma_bf16(a, b2, c[2]);
        c[3] = wmma_bf16(a, b3, c[3]);
    }
}

// Cooperative copy of this WG's 64-column weight slice (nt = wg_n*4..+3) into LDS.
__device__ __forceinline__ void stage_weights(const __bf16* __restrict__ gsrc,
                                              __bf16* ldst, int KT, int wg_n) {
    const int total_chunks = KT * 4 * 64;            // 16-byte chunks
    for (int c = threadIdx.x; c < total_chunks; c += blockDim.x) {
        int q    = c & 63;                           // chunk within fragment
        int fidx = c >> 6;                           // kt*4 + j
        int j    = fidx & 3;
        int kt   = fidx >> 2;
        int nt   = (wg_n << 2) + j;
        const v8bf* src = (const v8bf*)(gsrc + (((size_t)kt * 32 + nt) << 9)) + q;
        v8bf* dst = (v8bf*)(ldst + ((size_t)fidx << 9)) + q;
        *dst = *src;
    }
}

// Device-scope grid barrier (32 co-resident WGs; monotonically increasing gen).
__device__ __forceinline__ void grid_barrier(unsigned* cnt, unsigned* gen,
                                             unsigned nwg, unsigned myGen) {
    __syncthreads();
    __threadfence();                                    // release WG's global writes
    if (threadIdx.x == 0) {
        unsigned prev = __hip_atomic_fetch_add(cnt, 1u, __ATOMIC_ACQ_REL,
                                               __HIP_MEMORY_SCOPE_AGENT);
        if (prev == nwg - 1u) {
            __hip_atomic_store(cnt, 0u, __ATOMIC_RELAXED, __HIP_MEMORY_SCOPE_AGENT);
            __hip_atomic_fetch_add(gen, 1u, __ATOMIC_RELEASE, __HIP_MEMORY_SCOPE_AGENT);
        } else {
            while (__hip_atomic_load(gen, __ATOMIC_ACQUIRE,
                                     __HIP_MEMORY_SCOPE_AGENT) <= myGen) {
                __builtin_amdgcn_s_sleep(2);
            }
        }
    }
    __syncthreads();
    __threadfence();                                    // acquire others' writes
}

// ---------------- weight packing: fp32 row-major [K][N] -> WMMA B fragments ----------------
// Fragment element (kt, nt, lane, i) corresponds to
//   k = kt*32 + 8*(lane>>4) + (i<8 ? i : i+8),  n = nt*16 + (lane&15)
__global__ void pack_w_kernel(const float* __restrict__ W, __bf16* __restrict__ packed,
                              int K, int N) {
    const int NT = N >> 4;
    const size_t total = (size_t)K * N;
    for (size_t p = (size_t)blockIdx.x * blockDim.x + threadIdx.x; p < total;
         p += (size_t)gridDim.x * blockDim.x) {
        int i    = (int)(p & 15);
        int lane = (int)((p >> 4) & 31);
        size_t rest = p >> 9;
        int nt = (int)(rest % NT);
        int kt = (int)(rest / NT);
        int k = kt * 32 + ((lane >> 4) << 3) + (i < 8 ? i : i + 8);
        int n = nt * 16 + (lane & 15);
        packed[p] = to_bf16(W[(size_t)k * N + n]);
    }
}

__global__ void f2bf_kernel(const float* __restrict__ src, __bf16* __restrict__ dst, int n) {
    for (int i = blockIdx.x * blockDim.x + threadIdx.x; i < n; i += gridDim.x * blockDim.x)
        dst[i] = to_bf16(src[i]);
}

__global__ void init_kernel(__bf16* __restrict__ h0buf, __bf16* __restrict__ hs0,
                            unsigned* __restrict__ bar) {
    const __bf16 z = to_bf16(0.0f);
    const size_t n = (size_t)B_ * H_;
    for (size_t i = (size_t)blockIdx.x * blockDim.x + threadIdx.x; i < n;
         i += (size_t)gridDim.x * blockDim.x) {
        h0buf[i] = z;          // h0 ping buffer 0 (initial hidden state)
        hs0[i]   = z;          // hs_all step 0   (initial h1 state)
    }
    if (blockIdx.x == 0 && threadIdx.x == 0) { bar[0] = 0u; bar[1] = 0u; }
}

// ---------------- persistent recurrence kernel ----------------
// 32 WGs x 256 threads (8 waves). Wave tile 16(M) x 64(N) = 4 accumulators.
// WG tile 128 x 64. Per-WG weight slices (208 KB) staged once in LDS;
// B operands ds_load-fed, only the evolving hidden state comes from L2.
// Per step:
//   layer0: h0_new = tanh(emb[x[:,t]] @ Wih0 + h0_prev @ Whh0 + b0)   (K = 128 + 512)
//   <grid barrier>
//   layer1: hs[t+1] = tanh(h0_new @ Wih1 + hs[t] @ Whh1 + b1)         (K = 512 + 512)
//   <grid barrier>
__global__ void __launch_bounds__(256) rnn_scan_kernel(
    const int* __restrict__ x, const __bf16* __restrict__ embbf,
    const __bf16* __restrict__ pWih0, const __bf16* __restrict__ pWhh0,
    const __bf16* __restrict__ pWih1, const __bf16* __restrict__ pWhh1,
    const float* __restrict__ bih0, const float* __restrict__ bhh0,
    const float* __restrict__ bih1, const float* __restrict__ bhh1,
    __bf16* __restrict__ h0buf, __bf16* __restrict__ hs_all,
    unsigned* __restrict__ bar) {

    extern __shared__ __bf16 smem[];
    __bf16* sWih0 = smem + LDS_WIH0_OFF;
    __bf16* sWhh0 = smem + LDS_WHH0_OFF;
    __bf16* sWih1 = smem + LDS_WIH1_OFF;
    __bf16* sWhh1 = smem + LDS_WHH1_OFF;

    const int lane    = threadIdx.x & 31;
    const int wave    = threadIdx.x >> 5;
    const int wg_m    = blockIdx.x >> 3;           // 0..3
    const int wg_n    = blockIdx.x & 7;            // 0..7
    const int m0      = wg_m * 128 + wave * 16;    // wave row base
    const int n0      = wg_n * 64;                 // wave col base (4 x 16)
    const int half    = lane >> 4;
    const int collane = lane & 15;
    const int mlane   = m0 + collane;              // A-fragment row for this lane
    unsigned* cnt = bar;
    unsigned* gen = bar + 1;
    unsigned myGen = 0;

    // stage this WG's weight column-slices into LDS (once)
    stage_weights(pWih0, sWih0, E_ / 32, wg_n);
    stage_weights(pWhh0, sWhh0, H_ / 32, wg_n);
    stage_weights(pWih1, sWih1, H_ / 32, wg_n);
    stage_weights(pWhh1, sWhh1, H_ / 32, wg_n);

    // epilogue biases (registers, L2-broadcast loads once)
    float b0[4], b1[4];
#pragma unroll
    for (int j = 0; j < 4; ++j) {
        int col = n0 + j * 16 + collane;
        b0[j] = bih0[col] + bhh0[col];
        b1[j] = bih1[col] + bhh1[col];
    }
    __syncthreads();

    for (int t = 0; t < S_; ++t) {
        const __bf16* h0_prev = h0buf + (size_t)(t & 1) * (B_ * H_);
        __bf16*       h0_new  = h0buf + (size_t)((t + 1) & 1) * (B_ * H_);

        // ---------- layer 0 ----------
        v8f c[4];
#pragma unroll
        for (int j = 0; j < 4; ++j)
#pragma unroll
            for (int r = 0; r < 8; ++r) c[j][r] = 0.0f;

        // embedding gather contribution: A = emb[x[b, t]], K = E = 128
        const int tok = x[(size_t)mlane * S_ + t];
        gemm_acc_lds<E_ / 32>(embbf + (size_t)tok * E_, sWih0, lane, c);
        // recurrent contribution: A = h0_prev, K = H = 512
        gemm_acc_lds<H_ / 32>(h0_prev + (size_t)mlane * H_, sWhh0, lane, c);

        // epilogue: bias + tanh -> bf16 h0_new
#pragma unroll
        for (int j = 0; j < 4; ++j) {
            int col = n0 + j * 16 + collane;
#pragma unroll
            for (int r = 0; r < 8; ++r) {
                int row = m0 + half * 8 + r;
                h0_new[(size_t)row * H_ + col] = to_bf16(tanhf(c[j][r] + b0[j]));
            }
        }
        grid_barrier(cnt, gen, gridDim.x, myGen); ++myGen;

        // ---------- layer 1 ----------
        const __bf16* h1_prev = hs_all + (size_t)t * (B_ * H_);
        __bf16*       h1_new  = hs_all + (size_t)(t + 1) * (B_ * H_);
#pragma unroll
        for (int j = 0; j < 4; ++j)
#pragma unroll
            for (int r = 0; r < 8; ++r) c[j][r] = 0.0f;

        gemm_acc_lds<H_ / 32>(h0_new  + (size_t)mlane * H_, sWih1, lane, c);
        gemm_acc_lds<H_ / 32>(h1_prev + (size_t)mlane * H_, sWhh1, lane, c);

#pragma unroll
        for (int j = 0; j < 4; ++j) {
            int col = n0 + j * 16 + collane;
#pragma unroll
            for (int r = 0; r < 8; ++r) {
                int row = m0 + half * 8 + r;
                h1_new[(size_t)row * H_ + col] = to_bf16(tanhf(c[j][r] + b1[j]));
            }
        }
        grid_barrier(cnt, gen, gridDim.x, myGen); ++myGen;
    }
}

// ---------------- output projection: logits = hs @ Wout + bout ----------------
// A = hs_all[1..S] as [S*B][H] bf16 (row m = s*B + b), N = V = 128.
// Bandwidth-bound; global-fed WMMA GEMM. Grid 2048 WGs x 256 threads.
__global__ void __launch_bounds__(256) out_proj_kernel(
    const __bf16* __restrict__ hs, const __bf16* __restrict__ pWout,
    const float* __restrict__ bout, float* __restrict__ out) {

    const int lane    = threadIdx.x & 31;
    const int wave    = threadIdx.x >> 5;
    const int wg_m    = blockIdx.x >> 1;           // 0..1023
    const int wg_n    = blockIdx.x & 1;            // 0..1
    const int m0      = wg_m * 128 + wave * 16;
    const int n0      = wg_n * 64;
    const int half    = lane >> 4;
    const int collane = lane & 15;
    const int mlane   = m0 + collane;

    v8f c[4];
#pragma unroll
    for (int j = 0; j < 4; ++j)
#pragma unroll
        for (int r = 0; r < 8; ++r) c[j][r] = 0.0f;

    gemm_acc_glb<H_ / 32>(hs + (size_t)mlane * H_, pWout, V_ / 16, wg_n * 4, lane, c);

#pragma unroll
    for (int j = 0; j < 4; ++j) {
        int col = n0 + j * 16 + collane;
        float bb = bout[col];
#pragma unroll
        for (int r = 0; r < 8; ++r) {
            int m = m0 + half * 8 + r;       // m = s*B + b
            int s = m >> 9;                  // / B (=512)
            int b = m & (B_ - 1);
            out[((size_t)b * S_ + s) * V_ + col] = c[j][r] + bb;
        }
    }
}

// ---------------- host entry ----------------
extern "C" void kernel_launch(void* const* d_in, const int* in_sizes, int n_in,
                              void* d_out, int out_size, void* d_ws, size_t ws_size,
                              hipStream_t stream) {
    (void)in_sizes; (void)n_in; (void)out_size; (void)ws_size;

    const int*   x    = (const int*)d_in[0];
    const float* emb  = (const float*)d_in[1];
    const float* Wih0 = (const float*)d_in[2];
    const float* bih0 = (const float*)d_in[3];
    const float* Whh0 = (const float*)d_in[4];
    const float* bhh0 = (const float*)d_in[5];
    const float* Wih1 = (const float*)d_in[6];
    const float* bih1 = (const float*)d_in[7];
    const float* Whh1 = (const float*)d_in[8];
    const float* bhh1 = (const float*)d_in[9];
    const float* Wout = (const float*)d_in[10];
    const float* bout = (const float*)d_in[11];
    float* out = (float*)d_out;

    // workspace carve-up (all regions 256B aligned)
    char* ws = (char*)d_ws;
    size_t off = 0;
    auto take = [&](size_t nbytes) -> void* {
        void* p = (void*)(ws + off);
        off += (nbytes + 255) & ~(size_t)255;
        return p;
    };
    __bf16* hs_all = (__bf16*)take((size_t)(S_ + 1) * B_ * H_ * sizeof(__bf16)); // ~134.7 MB
    __bf16* h0buf  = (__bf16*)take((size_t)2 * B_ * H_ * sizeof(__bf16));
    __bf16* embbf  = (__bf16*)take((size_t)V_ * E_ * sizeof(__bf16));
    __bf16* pWih0  = (__bf16*)take((size_t)E_ * H_ * sizeof(__bf16));
    __bf16* pWhh0  = (__bf16*)take((size_t)H_ * H_ * sizeof(__bf16));
    __bf16* pWih1  = (__bf16*)take((size_t)H_ * H_ * sizeof(__bf16));
    __bf16* pWhh1  = (__bf16*)take((size_t)H_ * H_ * sizeof(__bf16));
    __bf16* pWout  = (__bf16*)take((size_t)H_ * V_ * sizeof(__bf16));
    unsigned* bar  = (unsigned*)take(256);

    // one-time (per call) conversions / packing — all tiny, L2-resident
    f2bf_kernel<<<32, 256, 0, stream>>>(emb, embbf, V_ * E_);
    pack_w_kernel<<<256, 256, 0, stream>>>(Wih0, pWih0, E_, H_);
    pack_w_kernel<<<256, 256, 0, stream>>>(Whh0, pWhh0, H_, H_);
    pack_w_kernel<<<256, 256, 0, stream>>>(Wih1, pWih1, H_, H_);
    pack_w_kernel<<<256, 256, 0, stream>>>(Whh1, pWhh1, H_, H_);
    pack_w_kernel<<<256, 256, 0, stream>>>(Wout, pWout, H_, V_);
    init_kernel<<<64, 256, 0, stream>>>(h0buf, hs_all, bar);

    // whole sequential scan in one persistent kernel (32 co-resident WGs,
    // 208 KB dynamic LDS each -> one WG per WGP)
    rnn_scan_kernel<<<32, 256, LDS_BYTES, stream>>>(x, embbf, pWih0, pWhh0, pWih1, pWhh1,
                                                    bih0, bhh0, bih1, bhh1,
                                                    h0buf, hs_all, bar);

    // output projection over all timesteps
    out_proj_kernel<<<2048, 256, 0, stream>>>(hs_all + (size_t)B_ * H_, pWout, bout, out);
}
